// GraphSAGE_89945205113354
// MI455X (gfx1250) — matrix operand order repaired and checked by reading, not compiled, not run on hardware
//
#include <hip/hip_runtime.h>
#include <cstdint>
#include <cstddef>

#define F 128        // F_IN = H = D = 128
#define NG 256       // num graphs (B)
#define BN_EPS 1e-5f

// ---------------------------------------------------------------------------
// Types for WMMA (wave32, gfx1250)
// ---------------------------------------------------------------------------
typedef __attribute__((ext_vector_type(16))) __bf16 v16bf;
typedef __attribute__((ext_vector_type(8)))  float  v8f;

union BF16x16 { uint4 q[2]; v16bf v; };

__device__ __forceinline__ unsigned short f2b(float f) {
  unsigned u = __float_as_uint(f);
  return (unsigned short)((u + 0x7fffu + ((u >> 16) & 1u)) >> 16);   // RNE
}
__device__ __forceinline__ float b2f(unsigned short s) {
  return __uint_as_float(((unsigned)s) << 16);
}
__device__ __forceinline__ float sigm(float x) { return 1.0f / (1.0f + expf(-x)); }

__device__ __forceinline__ void atomicMaxF(float* addr, float v) {
  if (v >= 0.0f) atomicMax((int*)addr, __float_as_int(v));
  else           atomicMin((unsigned int*)addr, __float_as_uint(v));
}

// ---------------------------------------------------------------------------
// Utility kernels
// ---------------------------------------------------------------------------
__global__ void k_cvt_f32_bf16(const float* __restrict__ in,
                               unsigned short* __restrict__ out, int n) {
  int i = blockIdx.x * blockDim.x + threadIdx.x;
  if (i < n) out[i] = f2b(in[i]);
}

__global__ void k_edge_count(const int* __restrict__ dst, float* __restrict__ cnt, int E) {
  int e = blockIdx.x * blockDim.x + threadIdx.x;
  if (e < E) atomicAdd(&cnt[dst[e]], 1.0f);
}

// agg[dst, f] += x[src, f]   (f32 source)
__global__ void k_scatter_f32(const float* __restrict__ x, const int* __restrict__ src,
                              const int* __restrict__ dst, float* __restrict__ agg, int total) {
  int i = blockIdx.x * blockDim.x + threadIdx.x;
  if (i >= total) return;
  int e = i >> 7, f = i & (F - 1);
  float v = x[(size_t)src[e] * F + f];
  atomicAdd(&agg[(size_t)dst[e] * F + f], v);
}

// agg[dst, f] += x[src, f]   (bf16 source)
__global__ void k_scatter_bf16(const unsigned short* __restrict__ x, const int* __restrict__ src,
                               const int* __restrict__ dst, float* __restrict__ agg, int total) {
  int i = blockIdx.x * blockDim.x + threadIdx.x;
  if (i >= total) return;
  int e = i >> 7, f = i & (F - 1);
  float v = b2f(x[(size_t)src[e] * F + f]);
  atomicAdd(&agg[(size_t)dst[e] * F + f], v);
}

// meanb = bf16(agg / max(cnt, 1))
__global__ void k_mean_cvt(const float* __restrict__ agg, const float* __restrict__ cnt,
                           unsigned short* __restrict__ meanb, int total) {
  int i = blockIdx.x * blockDim.x + threadIdx.x;
  if (i >= total) return;
  float c = cnt[i >> 7];
  c = c < 1.0f ? 1.0f : c;
  meanb[i] = f2b(agg[i] / c);
}

// ---------------------------------------------------------------------------
// Fused dual-GEMM + bias + BN(eval) + ReLU:
//   out = relu(bn(A1 @ B1^T + bias + A2 @ B2^T))
// A1,A2: [Nrows,128] bf16 row-major; B1,B2: [128,128] bf16 row-major [out,in].
// One wave per 16-row x 128-col strip: A fragments loaded once per K-chunk
// and reused across all 8 column tiles (8 f32 accumulators).
// 4 K-chunks x 8 col-tiles x 2 GEMMs = 64 v_wmma_f32_16x16x32_bf16 per wave.
// ---------------------------------------------------------------------------
__global__ void k_gemm_dual_bn_relu(
    const unsigned short* __restrict__ A1, const unsigned short* __restrict__ A2,
    const unsigned short* __restrict__ B1, const unsigned short* __restrict__ B2,
    const float* __restrict__ bias,
    const float* __restrict__ gam,  const float* __restrict__ bet,
    const float* __restrict__ rmean, const float* __restrict__ rvar,
    float* __restrict__ outF, unsigned short* __restrict__ outB, int nStrips) {
  const int strip = blockIdx.x * (blockDim.x >> 5) + (threadIdx.x >> 5);
  if (strip >= nStrips) return;                  // wave-uniform: EXEC stays all-1s
  const int lane = threadIdx.x & 31;
  const int half = lane >> 4;                    // 0 | 1
  const int l15  = lane & 15;
  const int row0 = strip << 4;

  const unsigned short* a1r = A1 + (size_t)(row0 + l15) * F;
  const unsigned short* a2r = A2 + (size_t)(row0 + l15) * F;
  const unsigned short* b1r = B1 + (size_t)l15 * F;   // column l15 of each tile
  const unsigned short* b2r = B2 + (size_t)l15 * F;
  __builtin_prefetch(b1r, 0, 1);                 // global_prefetch_b8
  __builtin_prefetch(b2r, 0, 1);

  v8f acc[8];
#pragma unroll
  for (int ct = 0; ct < 8; ++ct) acc[ct] = (v8f){0.f, 0.f, 0.f, 0.f, 0.f, 0.f, 0.f, 0.f};

#pragma unroll
  for (int kc = 0; kc < F; kc += 32) {
    BF16x16 a1, a2;
    // A layout: lane holds K = {kc+8h .. +7} then {kc+16+8h .. +7}
    a1.q[0] = *(const uint4*)(a1r + kc + half * 8);
    a1.q[1] = *(const uint4*)(a1r + kc + 16 + half * 8);
    a2.q[0] = *(const uint4*)(a2r + kc + half * 8);
    a2.q[1] = *(const uint4*)(a2r + kc + 16 + half * 8);
#pragma unroll
    for (int ct = 0; ct < 8; ++ct) {
      // B layout: lane holds K = {kc+8h .. +7} then {kc+16+8h .. +7} (column = ct*16 + l15)
      const size_t boff = (size_t)(ct * 16) * F + kc + half * 8;
      BF16x16 b1v, b2v;
      b1v.q[0] = *(const uint4*)(b1r + boff);
      b1v.q[1] = *(const uint4*)(b1r + boff + 16);
      b2v.q[0] = *(const uint4*)(b2r + boff);
      b2v.q[1] = *(const uint4*)(b2r + boff + 16);
      acc[ct] = __builtin_amdgcn_wmma_f32_16x16x32_bf16(false, a1.v, false, b1v.v,
                                                        (short)0, acc[ct], false, false);
      acc[ct] = __builtin_amdgcn_wmma_f32_16x16x32_bf16(false, a2.v, false, b2v.v,
                                                        (short)0, acc[ct], false, false);
    }
  }

  // Epilogue: per-lane column is ct*16 + l15; rows are r + 8*half.
#pragma unroll
  for (int ct = 0; ct < 8; ++ct) {
    const int col = ct * 16 + l15;
    const float bcol = bias[col];
    const float sc   = gam[col] * rsqrtf(rvar[col] + BN_EPS);
    const float rm   = rmean[col];
    const float bb   = bet[col];
#pragma unroll
    for (int r = 0; r < 8; ++r) {
      const int row = row0 + r + half * 8;
      float v = acc[ct][r] + bcol;
      v = (v - rm) * sc + bb;
      v = v > 0.0f ? v : 0.0f;
      if (outF) outF[(size_t)row * F + col] = v;
      if (outB) outB[(size_t)row * F + col] = f2b(v);
    }
  }
}

// ---------------------------------------------------------------------------
// Set2Set kernels
// ---------------------------------------------------------------------------
__global__ void k_lstm_gates(const float* __restrict__ qstar, const float* __restrict__ h,
                             const float* __restrict__ Wih, const float* __restrict__ Whh,
                             const float* __restrict__ bih, const float* __restrict__ bhh,
                             float* __restrict__ gates) {
  int i = blockIdx.x * blockDim.x + threadIdx.x;
  if (i >= NG * 4 * F) return;
  int b = i >> 9, j = i & (4 * F - 1);
  float s = bih[j] + bhh[j];
  const float4* q  = (const float4*)(qstar + (size_t)b * 2 * F);
  const float4* wi = (const float4*)(Wih + (size_t)j * 2 * F);
#pragma unroll 4
  for (int k = 0; k < 2 * F / 4; ++k) {
    float4 a = q[k], w = wi[k];
    s += a.x * w.x + a.y * w.y + a.z * w.z + a.w * w.w;
  }
  const float4* hh = (const float4*)(h + (size_t)b * F);
  const float4* wh = (const float4*)(Whh + (size_t)j * F);
#pragma unroll 4
  for (int k = 0; k < F / 4; ++k) {
    float4 a = hh[k], w = wh[k];
    s += a.x * w.x + a.y * w.y + a.z * w.z + a.w * w.w;
  }
  gates[i] = s;
}

__global__ void k_lstm_cell(const float* __restrict__ gates, float* __restrict__ h,
                            float* __restrict__ c, float* __restrict__ qstar) {
  int i = blockIdx.x * blockDim.x + threadIdx.x;
  if (i >= NG * F) return;
  int b = i >> 7, d = i & (F - 1);
  const float* g = gates + (size_t)b * 4 * F;
  float ig = sigm(g[d]);
  float fg = sigm(g[F + d]);
  float gg = tanhf(g[2 * F + d]);
  float og = sigm(g[3 * F + d]);
  float cn = fg * c[i] + ig * gg;
  float hn = og * tanhf(cn);
  c[i] = cn;
  h[i] = hn;
  qstar[(size_t)b * 2 * F + d] = hn;             // q-part of q_star
}

__global__ void k_s2s_init(float* __restrict__ qstar, float* __restrict__ emax,
                           float* __restrict__ denom) {
  int i = blockIdx.x * blockDim.x + threadIdx.x;
  if (i >= NG * F) return;
  int b = i >> 7, d = i & (F - 1);
  qstar[(size_t)b * 2 * F + F + d] = 0.0f;       // zero r-part
  if (d == 0) { emax[b] = __uint_as_float(0xff800000u); denom[b] = 0.0f; }
}

__global__ void k_s2s_e(const float* __restrict__ h2, const float* __restrict__ h,
                        const int* __restrict__ batch, float* __restrict__ e,
                        float* __restrict__ emax, int N) {
  int n = blockIdx.x * blockDim.x + threadIdx.x;
  if (n >= N) return;
  int b = batch[n];
  const float4* xr = (const float4*)(h2 + (size_t)n * F);
  const float4* qr = (const float4*)(h + (size_t)b * F);
  float s = 0.0f;
#pragma unroll 8
  for (int k = 0; k < F / 4; ++k) {
    float4 a = xr[k], q = qr[k];
    s += a.x * q.x + a.y * q.y + a.z * q.z + a.w * q.w;
  }
  e[n] = s;
  atomicMaxF(&emax[b], s);
}

__global__ void k_s2s_a(const float* __restrict__ e, const int* __restrict__ batch,
                        const float* __restrict__ emax, float* __restrict__ denom,
                        float* __restrict__ a, int N) {
  int n = blockIdx.x * blockDim.x + threadIdx.x;
  if (n >= N) return;
  int b = batch[n];
  float v = expf(e[n] - emax[b]);
  a[n] = v;
  atomicAdd(&denom[b], v);
}

__global__ void k_s2s_r(const float* __restrict__ h2, const int* __restrict__ batch,
                        const float* __restrict__ a, const float* __restrict__ denom,
                        float* __restrict__ qstar, int total) {
  int i = blockIdx.x * blockDim.x + threadIdx.x;
  if (i >= total) return;
  int n = i >> 7, f = i & (F - 1);
  int b = batch[n];
  float w = a[n] / denom[b];
  atomicAdd(&qstar[(size_t)b * 2 * F + F + f], w * h2[(size_t)n * F + f]);
}

__global__ void k_proj(const float* __restrict__ qstar, const float* __restrict__ Wp,
                       const float* __restrict__ bp, float* __restrict__ out) {
  int i = blockIdx.x * blockDim.x + threadIdx.x;
  if (i >= NG * F) return;
  int b = i >> 7, d = i & (F - 1);
  float s = bp[d];
  const float4* q = (const float4*)(qstar + (size_t)b * 2 * F);
  const float4* w = (const float4*)(Wp + (size_t)d * 2 * F);
#pragma unroll 8
  for (int k = 0; k < 2 * F / 4; ++k) {
    float4 a = q[k], ww = w[k];
    s += a.x * ww.x + a.y * ww.y + a.z * ww.z + a.w * ww.w;
  }
  out[i] = s;
}

// ---------------------------------------------------------------------------
// Host: orchestrate the whole pipeline on `stream` (graph-capture safe).
// ---------------------------------------------------------------------------
extern "C" void kernel_launch(void* const* d_in, const int* in_sizes, int n_in,
                              void* d_out, int out_size, void* d_ws, size_t ws_size,
                              hipStream_t stream) {
  const float* x     = (const float*)d_in[0];
  const int*   ei    = (const int*)d_in[1];
  const int*   batch = (const int*)d_in[2];
  const float* W1l = (const float*)d_in[3];
  const float* b1l = (const float*)d_in[4];
  const float* W1r = (const float*)d_in[5];
  const float* g1  = (const float*)d_in[6];
  const float* be1 = (const float*)d_in[7];
  const float* rm1 = (const float*)d_in[8];
  const float* rv1 = (const float*)d_in[9];
  const float* W2l = (const float*)d_in[10];
  const float* b2l = (const float*)d_in[11];
  const float* W2r = (const float*)d_in[12];
  const float* g2  = (const float*)d_in[13];
  const float* be2 = (const float*)d_in[14];
  const float* rm2 = (const float*)d_in[15];
  const float* rv2 = (const float*)d_in[16];
  const float* Wih = (const float*)d_in[17];
  const float* Whh = (const float*)d_in[18];
  const float* bih = (const float*)d_in[19];
  const float* bhh = (const float*)d_in[20];
  const float* Wp  = (const float*)d_in[21];
  const float* bp  = (const float*)d_in[22];

  const int N = in_sizes[0] / F;     // 40000 (multiple of 16)
  const int E = in_sizes[1] / 2;     // 640000
  const int* src = ei;
  const int* dst = ei + E;

  // ---- workspace carve (256B aligned) ----
  char* wsp = (char*)d_ws;
  auto carve = [&](size_t bytes) -> char* {
    char* p = wsp;
    wsp += (bytes + 255) & ~(size_t)255;
    return p;
  };
  unsigned short* xb    = (unsigned short*)carve((size_t)N * F * 2);
  unsigned short* h1b   = (unsigned short*)carve((size_t)N * F * 2);
  unsigned short* meanb = (unsigned short*)carve((size_t)N * F * 2);
  float* agg   = (float*)carve((size_t)N * F * 4);
  float* h2    = (float*)carve((size_t)N * F * 4);
  float* cnt   = (float*)carve((size_t)N * 4);
  float* ebuf  = (float*)carve((size_t)N * 4);
  float* abuf  = (float*)carve((size_t)N * 4);
  float* gates = (float*)carve((size_t)NG * 4 * F * 4);
  float* hS    = (float*)carve((size_t)NG * F * 4);
  float* cS    = (float*)carve((size_t)NG * F * 4);
  float* qstar = (float*)carve((size_t)NG * 2 * F * 4);
  float* emax  = (float*)carve((size_t)NG * 4);
  float* denom = (float*)carve((size_t)NG * 4);
  unsigned short* w1lb = (unsigned short*)carve((size_t)F * F * 2);
  unsigned short* w1rb = (unsigned short*)carve((size_t)F * F * 2);
  unsigned short* w2lb = (unsigned short*)carve((size_t)F * F * 2);
  unsigned short* w2rb = (unsigned short*)carve((size_t)F * F * 2);

  const int T = 256;
  const int NF = N * F;          // 5,120,000
  const int EF = E * F;          // 81,920,000 (< 2^31)
  const int nStrips = N / 16;    // 2500 row strips (each wave does 16x128)

  // ---- bf16 conversions ----
  k_cvt_f32_bf16<<<(NF + T - 1) / T, T, 0, stream>>>(x, xb, NF);
  k_cvt_f32_bf16<<<(F * F + T - 1) / T, T, 0, stream>>>(W1l, w1lb, F * F);
  k_cvt_f32_bf16<<<(F * F + T - 1) / T, T, 0, stream>>>(W1r, w1rb, F * F);
  k_cvt_f32_bf16<<<(F * F + T - 1) / T, T, 0, stream>>>(W2l, w2lb, F * F);
  k_cvt_f32_bf16<<<(F * F + T - 1) / T, T, 0, stream>>>(W2r, w2rb, F * F);

  // ---- degree counts (same graph for both layers) ----
  hipMemsetAsync(cnt, 0, (size_t)N * 4, stream);
  k_edge_count<<<(E + T - 1) / T, T, 0, stream>>>(dst, cnt, E);

  // ---- layer 1: scatter-mean, dual-GEMM + BN + ReLU -> h1 (bf16) ----
  hipMemsetAsync(agg, 0, (size_t)NF * 4, stream);
  k_scatter_f32<<<(EF + T - 1) / T, T, 0, stream>>>(x, src, dst, agg, EF);
  k_mean_cvt<<<(NF + T - 1) / T, T, 0, stream>>>(agg, cnt, meanb, NF);
  k_gemm_dual_bn_relu<<<(nStrips + 3) / 4, 128, 0, stream>>>(
      meanb, xb, w1lb, w1rb, b1l, g1, be1, rm1, rv1, nullptr, h1b, nStrips);

  // ---- layer 2 -> h2 (f32) ----
  hipMemsetAsync(agg, 0, (size_t)NF * 4, stream);
  k_scatter_bf16<<<(EF + T - 1) / T, T, 0, stream>>>(h1b, src, dst, agg, EF);
  k_mean_cvt<<<(NF + T - 1) / T, T, 0, stream>>>(agg, cnt, meanb, NF);
  k_gemm_dual_bn_relu<<<(nStrips + 3) / 4, 128, 0, stream>>>(
      meanb, h1b, w2lb, w2rb, b2l, g2, be2, rm2, rv2, h2, nullptr, nStrips);

  // ---- Set2Set (3 steps) ----
  hipMemsetAsync(hS, 0, (size_t)NG * F * 4, stream);
  hipMemsetAsync(cS, 0, (size_t)NG * F * 4, stream);
  hipMemsetAsync(qstar, 0, (size_t)NG * 2 * F * 4, stream);
  for (int step = 0; step < 3; ++step) {
    k_lstm_gates<<<(NG * 4 * F + T - 1) / T, T, 0, stream>>>(qstar, hS, Wih, Whh, bih, bhh, gates);
    k_lstm_cell<<<(NG * F + T - 1) / T, T, 0, stream>>>(gates, hS, cS, qstar);
    k_s2s_init<<<(NG * F + T - 1) / T, T, 0, stream>>>(qstar, emax, denom);
    k_s2s_e<<<(N + T - 1) / T, T, 0, stream>>>(h2, hS, batch, ebuf, emax, N);
    k_s2s_a<<<(N + T - 1) / T, T, 0, stream>>>(ebuf, batch, emax, denom, abuf, N);
    k_s2s_r<<<(NF + T - 1) / T, T, 0, stream>>>(h2, batch, abuf, denom, qstar, NF);
  }

  // ---- final projection ----
  k_proj<<<(NG * F + T - 1) / T, T, 0, stream>>>(qstar, Wp, bp, (float*)d_out);
}
